// TransformerBlock_16174846837027
// MI455X (gfx1250) — matrix operand order repaired and checked by reading, not compiled, not run on hardware
//
#include <hip/hip_runtime.h>

// ============================================================================
// Transformer block for MI455X (gfx1250, wave32, WMMA, async global->LDS).
// - GEMMs: v_wmma_f32_16x16x32_bf16, block tile 128x256, wave tile 64x64
//   (16 WMMA per 16 ds_load_b128 per K-step), double-buffered LDS fed by
//   global_load_async_to_lds_b128 (ASYNCcnt overlap of load and compute).
// - Weights pre-transposed to [N,K] bf16: every fragment = contiguous b128s.
// - Flash attention (online softmax, exp2 domain, sqrt(D)*log2e folded into
//   the Q projection); row stats via ds_swizzle SWAPX butterflies.
// Fixed shapes: B=2 N=2048 D=1024 H=16 E=64 F=4096. Workspace ~190 MB.
// ============================================================================

typedef __attribute__((ext_vector_type(16))) __bf16        v16bf;
typedef __attribute__((ext_vector_type(8)))  float         v8f;
typedef __attribute__((ext_vector_type(8)))  unsigned int  v8u;

#define DEV static __device__ __forceinline__

DEV unsigned short f2bf(float f) {            // f32 -> bf16 bits, RNE
  unsigned u = __builtin_bit_cast(unsigned, f);
  unsigned r = u + 0x7FFFu + ((u >> 16) & 1u);
  return (unsigned short)(r >> 16);
}

DEV float exp2_fast(float x) { return __builtin_amdgcn_exp2f(x); }  // v_exp_f32

DEV v8f wmma_bf16(v16bf a, v16bf b, v8f c) {
  return __builtin_amdgcn_wmma_f32_16x16x32_bf16(false, a, false, b,
                                                 (short)0, c, false, false);
}

// Butterfly exchange within 32 lanes via DS_SWIZZLE (immediate pattern,
// no index-VGPR setup unlike ds_bpermute).
template <int XORMASK>
DEV float swz_xor(float v) {
  int r = __builtin_amdgcn_ds_swizzle(__builtin_bit_cast(int, v),
                                      (XORMASK << 10) | 0x1f);
  return __builtin_bit_cast(float, r);
}

// 32-bit LDS offset of a generic pointer to __shared__.
DEV unsigned lds_addr(const void* p) { return (unsigned)(unsigned long long)p; }

// Async DMA: 16 bytes per lane, global -> LDS, tracked by ASYNCcnt.
DEV void async_copy_b128(const void* gptr, unsigned lds_off) {
  asm volatile("global_load_async_to_lds_b128 %0, %1, off"
               :: "v"(lds_off), "v"(gptr)
               : "memory");
}
DEV void wait_async0() { asm volatile("s_wait_asynccnt 0x0" ::: "memory"); }

// A-fragment (16x32 bf16), tile stored [m][k] (K contiguous):
// lane l: m = l&15, half = l>>4; VGPR j=0..3: K = 8*half+{2j,2j+1},
// j=4..7: K = 16+8*half+{..}. Two contiguous b128 LDS loads.
DEV v16bf afrag(const unsigned short* base, int stride, int lane) {
  int m = lane & 15, half = (lane >> 4) & 1;
  const unsigned short* row = base + m * stride;
  const unsigned* p0 = (const unsigned*)(row + half * 8);
  const unsigned* p1 = (const unsigned*)(row + 16 + half * 8);
  v8u t;
  t[0] = p0[0]; t[1] = p0[1]; t[2] = p0[2]; t[3] = p0[3];
  t[4] = p1[0]; t[5] = p1[1]; t[6] = p1[2]; t[7] = p1[3];
  return __builtin_bit_cast(v16bf, t);
}

// B-fragment (32x16 bf16), tile stored [n][k] (K contiguous):
// lane l: n = l&15 selects row; j=0..3: K = 8*half+{2j,2j+1},
// j=4..7: K = 16+8*half+{..} (same order as afrag).
DEV v16bf bfrag(const unsigned short* base, int stride, int koff, int lane) {
  int n = lane & 15, half = (lane >> 4) & 1;
  const unsigned* p0 = (const unsigned*)(base + n * stride + koff + half * 8);
  const unsigned* p1 = (const unsigned*)(base + n * stride + koff + 16 + half * 8);
  v8u t;
#pragma unroll
  for (int j = 0; j < 4; ++j) { t[j] = p0[j]; t[j + 4] = p1[j]; }
  return __builtin_bit_cast(v16bf, t);
}

// ---------------------------------------------------------------------------
// Fused convert + transpose: W[K,N] f32 -> Wt[N,K] bf16 (coalesced both sides).
__global__ __launch_bounds__(256) void cvt_transpose_bf16(
    const float* __restrict__ src, unsigned short* __restrict__ dst,
    int K, int N) {
  __shared__ unsigned short t[32][33];
  int k0 = blockIdx.y * 32, n0 = blockIdx.x * 32;
  int tx = threadIdx.x & 31, ty = threadIdx.x >> 5;
#pragma unroll
  for (int r = ty; r < 32; r += 8)
    t[r][tx] = f2bf(src[(size_t)(k0 + r) * N + n0 + tx]);
  __syncthreads();
#pragma unroll
  for (int r = ty; r < 32; r += 8)
    dst[(size_t)(n0 + r) * K + k0 + tx] = t[tx][r];
}

// Per-head V transpose: v[B*N, D] bf16 -> vt[B*H][64][2048] bf16.
__global__ __launch_bounds__(256) void transpose_v(
    const unsigned short* __restrict__ v, unsigned short* __restrict__ vt) {
  __shared__ unsigned short t[32][33];
  int bh = blockIdx.z, b = bh >> 4, h = bh & 15;
  int n0 = blockIdx.x * 32, e0 = blockIdx.y * 32;
  int tx = threadIdx.x & 31, ty = threadIdx.x >> 5;
  const unsigned short* src = v + ((size_t)b * 2048 + n0) * 1024 + h * 64 + e0;
#pragma unroll
  for (int r = ty; r < 32; r += 8) t[r][tx] = src[(size_t)r * 1024 + tx];
  __syncthreads();
  unsigned short* dst = vt + ((size_t)bh * 64 + e0) * 2048 + n0;
#pragma unroll
  for (int r = ty; r < 32; r += 8) dst[(size_t)r * 2048 + tx] = t[tx][r];
}

// LayerNorm over last dim D = VEC*1024 (+optional residual, +optional ReLU),
// float4-vectorized, values held in registers across both passes; bf16 out.
template <int VEC>
__global__ __launch_bounds__(256) void ln_kernel(
    const float* __restrict__ x, const float* __restrict__ res,
    const float* __restrict__ g, const float* __restrict__ bia,
    unsigned short* __restrict__ out, int relu) {
  const int D = VEC * 1024;
  int row = blockIdx.x, tid = threadIdx.x;
  const float* xr = x + (size_t)row * D;
  float4 v[VEC];
  float s = 0.f, ss = 0.f;
#pragma unroll
  for (int i = 0; i < VEC; ++i) {
    int c = tid * 4 + i * 1024;
    float4 a = *(const float4*)(xr + c);
    if (res) {
      float4 bvec = *(const float4*)(res + (size_t)row * D + c);
      a.x += bvec.x; a.y += bvec.y; a.z += bvec.z; a.w += bvec.w;
    }
    v[i] = a;
    s += a.x + a.y + a.z + a.w;
    ss += a.x * a.x + a.y * a.y + a.z * a.z + a.w * a.w;
  }
  for (int off = 16; off > 0; off >>= 1) {
    s += __shfl_down(s, off);
    ss += __shfl_down(ss, off);
  }
  __shared__ float sb[8], sb2[8];
  int lane = tid & 31, wv = tid >> 5;
  if (lane == 0) { sb[wv] = s; sb2[wv] = ss; }
  __syncthreads();
  float ts = 0.f, tss = 0.f;
#pragma unroll
  for (int i = 0; i < 8; ++i) { ts += sb[i]; tss += sb2[i]; }
  float mean = ts / (float)D;
  float var = tss / (float)D - mean * mean;
  float rstd = rsqrtf(var + 1e-5f);
#pragma unroll
  for (int i = 0; i < VEC; ++i) {
    int c = tid * 4 + i * 1024;
    float4 gg = *(const float4*)(g + c);
    float4 bb = *(const float4*)(bia + c);
    float4 a = v[i];
    float y0 = (a.x - mean) * rstd * gg.x + bb.x;
    float y1 = (a.y - mean) * rstd * gg.y + bb.y;
    float y2 = (a.z - mean) * rstd * gg.z + bb.z;
    float y3 = (a.w - mean) * rstd * gg.w + bb.w;
    if (relu) {
      y0 = fmaxf(y0, 0.f); y1 = fmaxf(y1, 0.f);
      y2 = fmaxf(y2, 0.f); y3 = fmaxf(y3, 0.f);
    }
    ushort4 o;
    o.x = f2bf(y0); o.y = f2bf(y1); o.z = f2bf(y2); o.w = f2bf(y3);
    *(ushort4*)(out + (size_t)row * D + c) = o;
  }
}

// ---------------------------------------------------------------------------
// WMMA GEMM: C[M,N] = scale * (A[M,K](bf16) * Bt[N,K]^T + bias).
// 256 threads = 8 waves (2x4), block tile 128x256, wave tile 64x64, K step 32.
#define TLDS 40                 // 32 + 8 pad, keeps b128 16B-aligned

template <bool F32OUT>
__global__ __launch_bounds__(256) void gemm_bf16(
    const unsigned short* __restrict__ A, const unsigned short* __restrict__ Bt,
    const float* __restrict__ bias, void* __restrict__ Cout,
    float scale, int M, int N, int K) {
  __shared__ unsigned short As[2][128 * TLDS];
  __shared__ unsigned short Bs[2][256 * TLDS];
  const unsigned ASZ = 128 * TLDS * 2, BSZ = 256 * TLDS * 2;
  int tid = threadIdx.x, lane = tid & 31, wvid = tid >> 5;
  int m_blk = blockIdx.y * 128, n_blk = blockIdx.x * 256;
  int wm = (wvid & 1) * 64, wn = (wvid >> 1) * 64;
  v8f acc[4][4] = {};

  int ra = tid >> 1, sa = tid & 1;   // A: 128 rows x 2 segments
  const unsigned short* Ap = A + (size_t)(m_blk + ra) * K + sa * 16;
  const unsigned short* Bp = Bt + (size_t)(n_blk + tid) * K;  // B: 256 rows
  unsigned a_dst = lds_addr(&As[0][ra * TLDS + sa * 16]);
  unsigned b_dst = lds_addr(&Bs[0][tid * TLDS]);

  async_copy_b128(Ap, a_dst);        // prologue: stage K-step 0 into buf 0
  async_copy_b128(Bp, b_dst);
  async_copy_b128(Bp + 16, b_dst + 32);
  wait_async0();
  __syncthreads();

  int nsteps = K >> 5;
  for (int s = 0; s < nsteps; ++s) {
    int cur = s & 1;
    if (s + 1 < nsteps) {            // overlap next stage with compute
      size_t ko = (size_t)(s + 1) * 32;
      unsigned nb = cur ^ 1;
      async_copy_b128(Ap + ko, a_dst + nb * ASZ);
      async_copy_b128(Bp + ko, b_dst + nb * BSZ);
      async_copy_b128(Bp + ko + 16, b_dst + nb * BSZ + 32);
    }
    const unsigned short* Ac = As[cur];
    const unsigned short* Bc = Bs[cur];
    v16bf af[4];
#pragma unroll
    for (int i = 0; i < 4; ++i)
      af[i] = afrag(&Ac[(wm + i * 16) * TLDS], TLDS, lane);
#pragma unroll
    for (int jn = 0; jn < 4; ++jn) {
      v16bf bf = bfrag(&Bc[(wn + jn * 16) * TLDS], TLDS, 0, lane);
#pragma unroll
      for (int i = 0; i < 4; ++i)
        acc[i][jn] = wmma_bf16(af[i], bf, acc[i][jn]);
    }
    wait_async0();
    __syncthreads();
  }

  int nl = lane & 15, half = (lane >> 4) & 1;
#pragma unroll
  for (int jn = 0; jn < 4; ++jn) {
    int col = n_blk + wn + jn * 16 + nl;
    float bb = bias[col];
#pragma unroll
    for (int i = 0; i < 4; ++i) {
      int row0 = m_blk + wm + i * 16 + half * 8;
#pragma unroll
      for (int r = 0; r < 8; ++r) {
        float v = (acc[i][jn][r] + bb) * scale;
        size_t idx = (size_t)(row0 + r) * N + col;
        if (F32OUT) ((float*)Cout)[idx] = v;
        else        ((unsigned short*)Cout)[idx] = f2bf(v);
      }
    }
  }
}

// ---------------------------------------------------------------------------
// Flash attention. One block per (b, h, 128-row q tile); 8 waves, each owns
// 16 q rows; KV swept in 32-wide chunks, async double-buffered.
// q holds sqrt(D)*log2e pre-scaled values -> softmax runs in exp2 domain.
__global__ __launch_bounds__(256) void attn_kernel(
    const unsigned short* __restrict__ Q, const unsigned short* __restrict__ Kg,
    const unsigned short* __restrict__ Vt, float* __restrict__ Out) {
  __shared__ unsigned short Qs[128 * 72];
  __shared__ unsigned short Ks[2][32 * 72];
  __shared__ unsigned short Vs[2][64 * TLDS];
  __shared__ unsigned short Ps[8][16 * TLDS];   // per-wave prob re-layout

  int tid = threadIdx.x, lane = tid & 31, wv = tid >> 5;
  int blk = blockIdx.x;
  int qblk = blk & 15;           // 2048/128 q tiles per head
  int bh = blk >> 4;             // 0..31
  int h = bh & 15, b = bh >> 4;
  size_t brow = (size_t)b * 2048;
  int hoff = h * 64;

  // per-thread async staging descriptors for the KV sweep
  unsigned dstb; const unsigned short* srcb; size_t cstep; unsigned bstride;
  if (tid < 128) {                       // K chunk: 32 rows x 64 e, row-major
    int row = tid >> 2, seg = tid & 3;
    dstb = lds_addr(&Ks[0][row * 72 + seg * 16]);
    srcb = Kg + (brow + row) * 1024 + hoff + seg * 16;
    cstep = (size_t)32 * 1024;
    bstride = 32 * 72 * 2;
  } else {                               // V chunk: 64 e rows x 32 kv (vt)
    int u = tid - 128, e = u >> 1, seg = u & 1;
    dstb = lds_addr(&Vs[0][e * TLDS + seg * 16]);
    srcb = Vt + ((size_t)bh * 64 + e) * 2048 + seg * 16;
    cstep = 32;
    bstride = 64 * TLDS * 2;
  }

  // prologue: stage q tile (128x64) + KV chunk 0, all async
#pragma unroll
  for (int rep = 0; rep < 2; ++rep) {
    int idx = rep * 256 + tid;
    int row = idx >> 2, seg = idx & 3;
    async_copy_b128(Q + (brow + qblk * 128 + row) * 1024 + hoff + seg * 16,
                    lds_addr(&Qs[row * 72 + seg * 16]));
  }
  async_copy_b128(srcb, dstb);
  wait_async0();
  __syncthreads();

  v16bf qa0 = afrag(&Qs[(wv * 16) * 72], 72, lane);        // E 0..31
  v16bf qa1 = afrag(&Qs[(wv * 16) * 72 + 32], 72, lane);   // E 32..63

  v8f O0 = {}, O1 = {}, O2 = {}, O3 = {};
  float m_i[8], l_i[8];
#pragma unroll
  for (int r = 0; r < 8; ++r) { m_i[r] = -1e30f; l_i[r] = 0.f; }

  for (int ci = 0; ci < 64; ++ci) {
    if (ci + 1 < 64)
      async_copy_b128(srcb + (size_t)(ci + 1) * cstep,
                      dstb + ((ci + 1) & 1) * bstride);
    const unsigned short* Kc = Ks[ci & 1];
    const unsigned short* Vc = Vs[ci & 1];

    // score tile 16x32 (already in log2 domain): B = k^T from Kc[kvcol][e]
    v8f s0 = {}, s1 = {};
#pragma unroll
    for (int ks = 0; ks < 2; ++ks) {
      v16bf b0 = bfrag(&Kc[0], 72, ks * 32, lane);
      v16bf b1 = bfrag(&Kc[16 * 72], 72, ks * 32, lane);
      s0 = wmma_bf16(ks ? qa1 : qa0, b0, s0);
      s1 = wmma_bf16(ks ? qa1 : qa0, b1, s1);
    }

    // online softmax (exp2); row (r, lane>>4), cols across 16-lane groups
    float corr[8];
#pragma unroll
    for (int r = 0; r < 8; ++r) {
      float x0 = s0[r], x1 = s1[r];
      float a = fmaxf(x0, x1);
      a = fmaxf(a, swz_xor<1>(a));
      a = fmaxf(a, swz_xor<2>(a));
      a = fmaxf(a, swz_xor<4>(a));
      a = fmaxf(a, swz_xor<8>(a));
      float mnew = fmaxf(m_i[r], a);
      corr[r] = exp2_fast(m_i[r] - mnew);
      m_i[r] = mnew;
      x0 = exp2_fast(x0 - mnew);
      x1 = exp2_fast(x1 - mnew);
      float rs = x0 + x1;
      rs += swz_xor<1>(rs);
      rs += swz_xor<2>(rs);
      rs += swz_xor<4>(rs);
      rs += swz_xor<8>(rs);
      l_i[r] = l_i[r] * corr[r] + rs;
      s0[r] = x0; s1[r] = x1;
    }
#pragma unroll
    for (int r = 0; r < 8; ++r) {
      O0[r] *= corr[r]; O1[r] *= corr[r]; O2[r] *= corr[r]; O3[r] *= corr[r];
    }

    // re-layout probs C->A via per-wave LDS scratch (DS in-order per wave)
    unsigned short* P = Ps[wv];
    int nl = lane & 15, half = (lane >> 4) & 1;
#pragma unroll
    for (int r = 0; r < 8; ++r) {
      int rr = r + 8 * half;
      P[rr * TLDS + nl] = f2bf(s0[r]);
      P[rr * TLDS + 16 + nl] = f2bf(s1[r]);
    }
    __builtin_amdgcn_wave_barrier();
    v16bf pf = afrag(P, TLDS, lane);

    // O += P (16x32) * V (32x64); V fragments from transposed tile Vc[e][kv]
    O0 = wmma_bf16(pf, bfrag(&Vc[0 * TLDS], TLDS, 0, lane), O0);
    O1 = wmma_bf16(pf, bfrag(&Vc[16 * TLDS], TLDS, 0, lane), O1);
    O2 = wmma_bf16(pf, bfrag(&Vc[32 * TLDS], TLDS, 0, lane), O2);
    O3 = wmma_bf16(pf, bfrag(&Vc[48 * TLDS], TLDS, 0, lane), O3);

    wait_async0();
    __syncthreads();
  }

  int nl = lane & 15, half = (lane >> 4) & 1;
#pragma unroll
  for (int r = 0; r < 8; ++r) {
    float inv = 1.0f / l_i[r];
    size_t row = brow + (size_t)qblk * 128 + wv * 16 + r + 8 * half;
    float* o = Out + row * 1024 + hoff;
    o[nl]      = O0[r] * inv;
    o[16 + nl] = O1[r] * inv;
    o[32 + nl] = O2[r] * inv;
    o[48 + nl] = O3[r] * inv;
  }
}

// ---------------------------------------------------------------------------
extern "C" void kernel_launch(void* const* d_in, const int* in_sizes, int n_in,
                              void* d_out, int out_size, void* d_ws, size_t ws_size,
                              hipStream_t stream) {
  (void)in_sizes; (void)n_in; (void)out_size; (void)ws_size;
  const float* x   = (const float*)d_in[0];
  const float* wq  = (const float*)d_in[1];
  const float* bq  = (const float*)d_in[2];
  const float* wk  = (const float*)d_in[3];
  const float* bk  = (const float*)d_in[4];
  const float* wvv = (const float*)d_in[5];
  const float* bv  = (const float*)d_in[6];
  const float* g1  = (const float*)d_in[7];
  const float* b1  = (const float*)d_in[8];
  const float* g2  = (const float*)d_in[9];
  const float* b2  = (const float*)d_in[10];
  const float* w1  = (const float*)d_in[11];
  const float* bw1 = (const float*)d_in[12];
  const float* gf  = (const float*)d_in[13];
  const float* bfv = (const float*)d_in[14];
  const float* w2  = (const float*)d_in[15];
  const float* bw2 = (const float*)d_in[16];
  float* out = (float*)d_out;

  const int BNr = 4096;             // B*N rows
  const int D = 1024, F = 4096;
  const float QSCALE = 46.166241308446828f;  // sqrt(D) * log2(e)

  char* ws = (char*)d_ws;
  size_t off = 0;
  auto alloc = [&](size_t bytes) -> void* {
    void* p = ws + off;
    off += (bytes + 255) & ~(size_t)255;
    return p;
  };
  unsigned short* h_bf  = (unsigned short*)alloc((size_t)BNr * D * 2);
  unsigned short* q_bf  = (unsigned short*)alloc((size_t)BNr * D * 2);
  unsigned short* k_bf  = (unsigned short*)alloc((size_t)BNr * D * 2);
  unsigned short* v_bf  = (unsigned short*)alloc((size_t)BNr * D * 2);
  unsigned short* vt_bf = (unsigned short*)alloc((size_t)BNr * D * 2);
  unsigned short* h2_bf = (unsigned short*)alloc((size_t)BNr * D * 2);
  unsigned short* u_bf  = (unsigned short*)alloc((size_t)BNr * F * 2);
  float*          attn  = (float*)alloc((size_t)BNr * D * 4);
  float*          u_pre = (float*)alloc((size_t)BNr * F * 4);
  unsigned short* wqt = (unsigned short*)alloc((size_t)D * D * 2);
  unsigned short* wkt = (unsigned short*)alloc((size_t)D * D * 2);
  unsigned short* wvt = (unsigned short*)alloc((size_t)D * D * 2);
  unsigned short* w1t = (unsigned short*)alloc((size_t)D * F * 2);
  unsigned short* w2t = (unsigned short*)alloc((size_t)F * D * 2);

  dim3 blk(256);
  // weights: convert + transpose to [N,K] bf16
  cvt_transpose_bf16<<<dim3(D / 32, D / 32), blk, 0, stream>>>(wq, wqt, D, D);
  cvt_transpose_bf16<<<dim3(D / 32, D / 32), blk, 0, stream>>>(wk, wkt, D, D);
  cvt_transpose_bf16<<<dim3(D / 32, D / 32), blk, 0, stream>>>(wvv, wvt, D, D);
  cvt_transpose_bf16<<<dim3(F / 32, D / 32), blk, 0, stream>>>(w1, w1t, D, F);
  cvt_transpose_bf16<<<dim3(D / 32, F / 32), blk, 0, stream>>>(w2, w2t, F, D);

  // h = LN(x)
  ln_kernel<1><<<BNr, blk, 0, stream>>>(x, nullptr, g1, b1, h_bf, 0);
  // q,k,v = h @ W + b  (q pre-scaled by sqrt(D)*log2e for exp2 softmax)
  gemm_bf16<false><<<dim3(D / 256, BNr / 128), blk, 0, stream>>>(h_bf, wqt, bq, q_bf, QSCALE, BNr, D, D);
  gemm_bf16<false><<<dim3(D / 256, BNr / 128), blk, 0, stream>>>(h_bf, wkt, bk, k_bf, 1.0f, BNr, D, D);
  gemm_bf16<false><<<dim3(D / 256, BNr / 128), blk, 0, stream>>>(h_bf, wvt, bv, v_bf, 1.0f, BNr, D, D);
  // per-head V transpose for the P*V WMMA
  transpose_v<<<dim3(64, 2, 32), blk, 0, stream>>>(v_bf, vt_bf);
  // flash attention: 32 heads * 16 q-tiles
  attn_kernel<<<512, blk, 0, stream>>>(q_bf, k_bf, vt_bf, attn);
  // h2 = LN(x + attn)
  ln_kernel<1><<<BNr, blk, 0, stream>>>(x, attn, g2, b2, h2_bf, 0);
  // u_pre = h2 @ w1 + bw1  (f32 out: LN over F follows)
  gemm_bf16<true><<<dim3(F / 256, BNr / 128), blk, 0, stream>>>(h2_bf, w1t, bw1, u_pre, 1.0f, BNr, F, D);
  // u = relu(LN(u_pre))  -> bf16
  ln_kernel<4><<<BNr, blk, 0, stream>>>(u_pre, nullptr, gf, bfv, u_bf, 1);
  // out = u @ w2 + bw2   (f32 final)
  gemm_bf16<true><<<dim3(D / 256, BNr / 128), blk, 0, stream>>>(u_bf, w2t, bw2, out, 1.0f, BNr, D, F);
}
